// SpikeICSPB3DLMMinimal_35373350650270
// MI455X (gfx1250) — matrix operand order repaired and checked by reading, not compiled, not run on hardware
//
#include <hip/hip_runtime.h>
#include <hip/hip_bf16.h>

typedef __bf16 bf16_t;
typedef __attribute__((ext_vector_type(16))) __bf16 v16bf;
typedef __attribute__((ext_vector_type(8)))  __bf16 v8bf;
typedef __attribute__((ext_vector_type(8)))  float  v8f;

#define B_ 16
#define T_ 512
#define H_ 512
#define P_ 512
#define V_ 8192
#define PH_ 16
#define TOPK_ 6
#define RADIUS_ 0.42f
#define BSCALE_ 0.18f
#define LIMIT_ 1.25f

enum { ACT_NONE = 0, ACT_GELU = 1, ACT_TANH = 2 };

__device__ __forceinline__ float gelu_exact(float x) {
  return 0.5f * x * (1.0f + erff(x * 0.7071067811865475f));
}

// ---------------------------------------------------------------------------
// Generic 16-row WMMA linear layer: C[16,N] = act(A[16,K] @ W[N,K]^T + bias)
// A is given as up to four 512-wide bf16 segments (concat without copies).
// One wave computes one 16x16 tile of C (M=16 == batch). DUAL=true adds a
// second weight matrix W2 that reuses the A fragment (sel + brl fusion).
// The segment loop is fully unrolled with a ternary pointer chain so every
// A access references a kernel argument directly -> global_load_b128 (no
// flat loads / no DScnt coupling in the hot loop).
// ---------------------------------------------------------------------------
template <bool DUAL>
__global__ void lin16_wmma(const bf16_t* __restrict__ A0, const bf16_t* __restrict__ A1,
                           const bf16_t* __restrict__ A2, const bf16_t* __restrict__ A3,
                           const bf16_t* __restrict__ W, const float* __restrict__ bias,
                           float* __restrict__ Cf, bf16_t* __restrict__ Cbf,
                           int K, int N, int act, long cstride,
                           const bf16_t* __restrict__ W2, const float* __restrict__ bias2,
                           float* __restrict__ Cf2)
{
  const int lane = threadIdx.x & 31;
  const int wave = threadIdx.x >> 5;
  const int tile = blockIdx.x * (blockDim.x >> 5) + wave;
  const int n0 = tile * 16;
  if (n0 >= N) return;  // wave-uniform; grids are sized exactly

  const int row = lane & 15;   // A-matrix M index (both lane halves)
  const int kh  = lane >> 4;   // which K-half this lane holds
  const int col = lane & 15;   // B/C column

  v8f acc  = {};
  v8f acc2 = {};
  const int nseg = K >> 9;     // number of 512-wide concat segments
  const size_t wrow = (size_t)(n0 + col) * K + kh * 16;

#pragma unroll
  for (int sg = 0; sg < 4; ++sg) {
    if (sg < nseg) {
      const bf16_t* Ap = (sg == 0) ? A0 : (sg == 1) ? A1 : (sg == 2) ? A2 : A3;
      const bf16_t* Wp  = W + wrow + (sg << 9);
      const bf16_t* Wp2 = DUAL ? (W2 + wrow + (sg << 9)) : nullptr;
      const bf16_t* Arow = Ap + row * 512 + kh * 16;
#pragma unroll 4
      for (int kc = 0; kc < 16; ++kc) {
        const int kl = kc << 5;
        // A fragment: 16x32 bf16; lane half kh holds K {kh*16..+15}
        const v8bf lo = *(const v8bf*)(Arow + kl);
        const v8bf hi = *(const v8bf*)(Arow + kl + 8);
        v16bf a;
#pragma unroll
        for (int i = 0; i < 8; ++i) { a[i] = lo[i]; a[i + 8] = hi[i]; }
        // B fragment: 32x16 bf16 from row-major W[N,K]
        const v16bf b = *(const v16bf*)(Wp + kl);
        __builtin_prefetch(Wp + kl + 64, 0, 1);   // stream next chunks (L2-resident)
        acc = __builtin_amdgcn_wmma_f32_16x16x32_bf16(
            false, a, false, b, (short)0, acc, false, false);
        if (DUAL) {
          const v16bf b2 = *(const v16bf*)(Wp2 + kl);
          acc2 = __builtin_amdgcn_wmma_f32_16x16x32_bf16(
              false, a, false, b2, (short)0, acc2, false, false);
        }
      }
    }
  }

  // C layout: VGPR r -> row r (lanes 0-15) / row r+8 (lanes 16-31)
  const int rbase = kh << 3;
#pragma unroll
  for (int r = 0; r < 8; ++r) {
    const int rr = rbase + r;
    const int cc = n0 + col;
    float v = acc[r] + (bias ? bias[cc] : 0.0f);
    if (act == ACT_GELU)      v = gelu_exact(v);
    else if (act == ACT_TANH) v = tanhf(v);
    if (Cf)  Cf[(long)rr * cstride + cc] = v;
    if (Cbf) Cbf[rr * N + cc] = (bf16_t)v;
    if (DUAL) {
      const float v2 = acc2[r] + (bias2 ? bias2[cc] : 0.0f);
      Cf2[(long)rr * cstride + cc] = v2;
    }
  }
}

// ---------------------------------------------------------------------------
// Block reductions (blockDim.x == 512)
// ---------------------------------------------------------------------------
__device__ __forceinline__ float red_sum512(float v, float* s) {
  s[threadIdx.x] = v; __syncthreads();
  for (int o = 256; o > 0; o >>= 1) {
    if (threadIdx.x < (unsigned)o) s[threadIdx.x] += s[threadIdx.x + o];
    __syncthreads();
  }
  const float r = s[0]; __syncthreads(); return r;
}
__device__ __forceinline__ float red_max512(float v, float* s) {
  s[threadIdx.x] = v; __syncthreads();
  for (int o = 256; o > 0; o >>= 1) {
    if (threadIdx.x < (unsigned)o) s[threadIdx.x] = fmaxf(s[threadIdx.x], s[threadIdx.x + o]);
    __syncthreads();
  }
  const float r = s[0]; __syncthreads(); return r;
}
__device__ __forceinline__ void red_argmax512(float v, float* s, int* si,
                                              float* ov, int* oi) {
  s[threadIdx.x] = v; si[threadIdx.x] = threadIdx.x; __syncthreads();
  for (int o = 256; o > 0; o >>= 1) {
    if (threadIdx.x < (unsigned)o) {
      if (s[threadIdx.x + o] > s[threadIdx.x]) {
        s[threadIdx.x]  = s[threadIdx.x + o];
        si[threadIdx.x] = si[threadIdx.x + o];
      }
    }
    __syncthreads();
  }
  *ov = s[0]; *oi = si[0]; __syncthreads();
}

// ---------------------------------------------------------------------------
// Mixer: softmax(sel)->bw, anchor, distances, local mask, top-k bridge,
// normalized mixture w, masses, mean distance, topo. One block per batch row.
// ---------------------------------------------------------------------------
__global__ void mixer_kernel(const float* __restrict__ sel, const float* __restrict__ brl,
                             const float* __restrict__ pos, const float* __restrict__ Wtp,
                             const float* __restrict__ btp,
                             float* __restrict__ out_w, float* __restrict__ out_lm,
                             float* __restrict__ out_bm, float* __restrict__ out_md,
                             bf16_t* __restrict__ w_bf, bf16_t* __restrict__ topo_bf,
                             int t)
{
  __shared__ float s[512];
  __shared__ int   si[512];
  __shared__ float s_tv[TOPK_];
  __shared__ float s_sc[2];

  const int b = blockIdx.x;
  const int p = threadIdx.x;

  // softmax over selector logits
  const float sv = sel[b * P_ + p];
  const float mx = red_max512(sv, s);
  const float e  = expf(sv - mx);
  const float se = red_sum512(e, s);
  const float bw = e / se;

  // anchor and distances
  const float px = pos[p * 3 + 0], py = pos[p * 3 + 1], pz = pos[p * 3 + 2];
  const float ax = red_sum512(bw * px, s);
  const float ay = red_sum512(bw * py, s);
  const float az = red_sum512(bw * pz, s);
  const float dx = ax - px, dy = ay - py, dz = az - pz;
  const float d  = sqrtf(dx * dx + dy * dy + dz * dz);
  const float lk = expf(-(d * d) / (2.0f * RADIUS_ * RADIUS_));

  float amv; int ami;
  red_argmax512(-d, s, si, &amv, &ami);
  const bool  lm  = (d <= RADIUS_) || (p == ami);
  const float lmf = lm ? 1.0f : 0.0f;

  // bridge logits with local mask; iterative top-k argmax
  float cur = lm ? -1e9f : brl[b * P_ + p];
  int rank = -1;
  for (int j = 0; j < TOPK_; ++j) {
    float wv; int wi;
    red_argmax512(cur, s, si, &wv, &wi);
    if (threadIdx.x == 0) s_tv[j] = wv;
    if (p == wi) { rank = j; cur = -3.0e38f; }
    __syncthreads();
  }
  if (threadIdx.x == 0) {
    const float m0 = s_tv[0];
    float den = 0.0f;
    for (int j = 0; j < TOPK_; ++j) den += expf(s_tv[j] - m0);
    s_sc[0] = m0; s_sc[1] = den;
  }
  __syncthreads();
  const float sp = (rank >= 0) ? expf(s_tv[rank] - s_sc[0]) / s_sc[1] : 0.0f;

  // mixture + normalization
  const float mixed = bw * lk * lmf + BSCALE_ * sp;
  const float msum  = red_sum512(mixed, s);
  const float wgt   = mixed / fmaxf(msum, 1e-6f);

  const long obase = (long)b * T_ + t;
  out_w[obase * P_ + p] = wgt;
  w_bf[b * P_ + p] = (bf16_t)wgt;

  const float lmass = red_sum512(wgt * lmf, s);
  const float bmass = red_sum512(wgt * (1.0f - lmf), s);
  const float mdist = red_sum512(wgt * d, s);
  if (threadIdx.x == 0) {
    out_lm[obase] = lmass; out_bm[obase] = bmass; out_md[obase] = mdist;
  }

  // topo = tanh([anchor, lmass, bmass] @ Wtp^T + btp), K=5 -> plain VALU
  const int h = p;
  const float tv = Wtp[h * 5 + 0] * ax + Wtp[h * 5 + 1] * ay + Wtp[h * 5 + 2] * az +
                   Wtp[h * 5 + 3] * lmass + Wtp[h * 5 + 4] * bmass + btp[h];
  topo_bf[b * H_ + h] = (bf16_t)tanhf(tv);
}

// ---------------------------------------------------------------------------
// Phase router: pw = softmax([sec|prev] @ Wpr^T + bpr), ps = pw @ phase_embed
// Tiny (16x16): single 256-thread block, f32 VALU.
// ---------------------------------------------------------------------------
__global__ void phase_kernel(const float* __restrict__ sec_f, const float* __restrict__ prev_f,
                             const float* __restrict__ Wpr, const float* __restrict__ bpr,
                             const float* __restrict__ phase_embed,
                             float* __restrict__ out_pw, bf16_t* __restrict__ ps_bf, int t)
{
  __shared__ float spw[B_][PH_];
  const int tid = threadIdx.x;
  const int b = tid >> 4, ph = tid & 15;
  float acc = bpr[ph];
  for (int k = 0; k < H_; ++k) acc += sec_f[b * H_ + k]  * Wpr[ph * (2 * H_) + k];
  for (int k = 0; k < H_; ++k) acc += prev_f[b * H_ + k] * Wpr[ph * (2 * H_) + H_ + k];
  spw[b][ph] = acc;
  __syncthreads();
  float mxv = -3.0e38f;
  for (int j = 0; j < PH_; ++j) mxv = fmaxf(mxv, spw[b][j]);
  float den = 0.0f;
  for (int j = 0; j < PH_; ++j) den += expf(spw[b][j] - mxv);
  const float pw = expf(acc - mxv) / den;
  __syncthreads();
  spw[b][ph] = pw;
  out_pw[((long)b * T_ + t) * PH_ + ph] = pw;
  __syncthreads();
  for (int i = tid; i < B_ * H_; i += 256) {
    const int bb = i >> 9, h = i & 511;
    float a = 0.0f;
    for (int j = 0; j < PH_; ++j) a += spw[bb][j] * phase_embed[j * H_ + h];
    ps_bf[i] = (bf16_t)a;
  }
}

// ---------------------------------------------------------------------------
// Gate + state update + LayerNorm + clip. One block per batch row.
// ---------------------------------------------------------------------------
__global__ void gate_hid_kernel(const float* __restrict__ g1_f, const float* __restrict__ su_f,
                                const float* __restrict__ Wpg2, const float* __restrict__ bpg2,
                                const float* __restrict__ ln_g, const float* __restrict__ ln_b,
                                float* __restrict__ prev_f, bf16_t* __restrict__ prev_bf,
                                bf16_t* __restrict__ hid_bf,
                                float* __restrict__ out_hid, float* __restrict__ out_gate, int t)
{
  __shared__ float s[512];
  const int b = blockIdx.x, h = threadIdx.x;
  const float g   = g1_f[b * H_ + h];              // already gelu'ed by lin16
  const float dot = red_sum512(g * Wpg2[h], s);
  const float gate = 1.0f / (1.0f + expf(-(dot + bpg2[0])));
  const float su = su_f[b * H_ + h];
  const float pv = prev_f[b * H_ + h];
  const float hid = gate * su + (1.0f - gate) * pv;
  const float mu  = red_sum512(hid, s) * (1.0f / H_);
  const float df  = hid - mu;
  const float var = red_sum512(df * df, s) * (1.0f / H_);
  float hn = df * rsqrtf(var + 1e-5f) * ln_g[h] + ln_b[h];
  hn = fminf(fmaxf(hn, -LIMIT_), LIMIT_);
  const long obase = (long)b * T_ + t;
  out_hid[obase * H_ + h] = hn;
  prev_f[b * H_ + h] = hn;
  prev_bf[b * H_ + h] = (bf16_t)hn;
  hid_bf[b * H_ + h]  = (bf16_t)hn;
  if (h == 0) out_gate[obase] = gate;
}

// ---------------------------------------------------------------------------
// One-time helpers
// ---------------------------------------------------------------------------
__global__ void f2bf_kernel(const float* __restrict__ src, bf16_t* __restrict__ dst, int n) {
  const int i = blockIdx.x * blockDim.x + threadIdx.x;
  if (i < n) dst[i] = (bf16_t)src[i];
}
__global__ void embed_kernel(const int* __restrict__ ids, const float* __restrict__ emb,
                             bf16_t* __restrict__ toks_bf) {
  const int i = blockIdx.x * blockDim.x + threadIdx.x;   // over T*B*H
  if (i >= T_ * B_ * H_) return;
  const int t = i / (B_ * H_);
  const int r = i - t * (B_ * H_);
  const int b = r >> 9;
  const int h = r & 511;
  const int id = ids[b * T_ + t];
  toks_bf[i] = (bf16_t)emb[(long)id * H_ + h];
}
__global__ void zero_state_kernel(float* __restrict__ pf, bf16_t* __restrict__ pb, int n) {
  const int i = blockIdx.x * blockDim.x + threadIdx.x;
  if (i < n) { pf[i] = 0.0f; pb[i] = (bf16_t)0.0f; }
}

// ---------------------------------------------------------------------------
extern "C" void kernel_launch(void* const* d_in, const int* in_sizes, int n_in,
                              void* d_out, int out_size, void* d_ws, size_t ws_size,
                              hipStream_t stream) {
  (void)in_sizes; (void)n_in; (void)out_size; (void)ws_size;
  const int*   ids         = (const int*)  d_in[0];
  const float* emb         = (const float*)d_in[1];
  const float* positions   = (const float*)d_in[2];
  const float* Wsel        = (const float*)d_in[3];
  const float* bsel        = (const float*)d_in[4];
  const float* Wbr         = (const float*)d_in[5];
  const float* bbr         = (const float*)d_in[6];
  const float* patch_vals  = (const float*)d_in[7];
  const float* Wsb1        = (const float*)d_in[8];
  const float* bsb1        = (const float*)d_in[9];
  const float* Wsb2        = (const float*)d_in[10];
  const float* bsb2        = (const float*)d_in[11];
  const float* phase_embed = (const float*)d_in[12];
  const float* Wpr         = (const float*)d_in[13];
  const float* bpr         = (const float*)d_in[14];
  const float* Wsu1        = (const float*)d_in[15];
  const float* bsu1        = (const float*)d_in[16];
  const float* Wsu2        = (const float*)d_in[17];
  const float* bsu2        = (const float*)d_in[18];
  const float* Wpg1        = (const float*)d_in[19];
  const float* bpg1        = (const float*)d_in[20];
  const float* Wpg2        = (const float*)d_in[21];
  const float* bpg2        = (const float*)d_in[22];
  const float* Wtp         = (const float*)d_in[23];
  const float* btp         = (const float*)d_in[24];
  const float* ln_g        = (const float*)d_in[25];
  const float* ln_b        = (const float*)d_in[26];
  const float* Wout        = (const float*)d_in[27];

  float* out = (float*)d_out;
  float* out_logits = out;
  float* out_hid  = out_logits + (size_t)B_ * T_ * V_;
  float* out_w    = out_hid    + (size_t)B_ * T_ * H_;
  float* out_pw   = out_w      + (size_t)B_ * T_ * P_;
  float* out_gate = out_pw     + (size_t)B_ * T_ * PH_;
  float* out_lm   = out_gate   + (size_t)B_ * T_;
  float* out_bm   = out_lm     + (size_t)B_ * T_;
  float* out_md   = out_bm     + (size_t)B_ * T_;

  // workspace carve-out (256B aligned)
  char* wsb = (char*)d_ws;
  size_t off = 0;
  auto alloc = [&](size_t bytes) -> char* {
    char* p = wsb + off;
    off = (off + bytes + 255) & ~(size_t)255;
    return p;
  };
  bf16_t* Wsel_bf = (bf16_t*)alloc((size_t)P_ * 2 * H_ * 2);
  bf16_t* Wbr_bf  = (bf16_t*)alloc((size_t)P_ * 2 * H_ * 2);
  bf16_t* pv_bf   = (bf16_t*)alloc((size_t)P_ * H_ * 2);
  bf16_t* Wsb1_bf = (bf16_t*)alloc((size_t)H_ * 3 * H_ * 2);
  bf16_t* Wsb2_bf = (bf16_t*)alloc((size_t)H_ * H_ * 2);
  bf16_t* Wsu1_bf = (bf16_t*)alloc((size_t)H_ * 4 * H_ * 2);
  bf16_t* Wsu2_bf = (bf16_t*)alloc((size_t)H_ * H_ * 2);
  bf16_t* Wpg1_bf = (bf16_t*)alloc((size_t)H_ * 2 * H_ * 2);
  bf16_t* Wout_bf = (bf16_t*)alloc((size_t)V_ * H_ * 2);
  bf16_t* toks_bf = (bf16_t*)alloc((size_t)T_ * B_ * H_ * 2);
  float*  prev_f  = (float*) alloc((size_t)B_ * H_ * 4);
  bf16_t* prev_bf = (bf16_t*)alloc((size_t)B_ * H_ * 2);
  float*  sel_f   = (float*) alloc((size_t)B_ * P_ * 4);
  float*  brl_f   = (float*) alloc((size_t)B_ * P_ * 4);
  bf16_t* w_bf    = (bf16_t*)alloc((size_t)B_ * P_ * 2);
  bf16_t* patch_bf= (bf16_t*)alloc((size_t)B_ * H_ * 2);
  bf16_t* t1_bf   = (bf16_t*)alloc((size_t)B_ * H_ * 2);
  float*  sec_f   = (float*) alloc((size_t)B_ * H_ * 4);
  bf16_t* sec_bf  = (bf16_t*)alloc((size_t)B_ * H_ * 2);
  bf16_t* ps_bf   = (bf16_t*)alloc((size_t)B_ * H_ * 2);
  bf16_t* topo_bf = (bf16_t*)alloc((size_t)B_ * H_ * 2);
  bf16_t* t2_bf   = (bf16_t*)alloc((size_t)B_ * H_ * 2);
  float*  su_f    = (float*) alloc((size_t)B_ * H_ * 4);
  bf16_t* su_bf   = (bf16_t*)alloc((size_t)B_ * H_ * 2);
  float*  g1_f    = (float*) alloc((size_t)B_ * H_ * 4);
  bf16_t* hid_bf  = (bf16_t*)alloc((size_t)B_ * H_ * 2);

  auto conv = [&](const float* s, bf16_t* d, int n) {
    f2bf_kernel<<<(n + 255) / 256, 256, 0, stream>>>(s, d, n);
  };
  conv(Wsel, Wsel_bf, P_ * 2 * H_);
  conv(Wbr,  Wbr_bf,  P_ * 2 * H_);
  conv(patch_vals, pv_bf, P_ * H_);
  conv(Wsb1, Wsb1_bf, H_ * 3 * H_);
  conv(Wsb2, Wsb2_bf, H_ * H_);
  conv(Wsu1, Wsu1_bf, H_ * 4 * H_);
  conv(Wsu2, Wsu2_bf, H_ * H_);
  conv(Wpg1, Wpg1_bf, H_ * 2 * H_);
  conv(Wout, Wout_bf, V_ * H_);

  embed_kernel<<<(T_ * B_ * H_ + 255) / 256, 256, 0, stream>>>(ids, emb, toks_bf);
  zero_state_kernel<<<(B_ * H_ + 255) / 256, 256, 0, stream>>>(prev_f, prev_bf, B_ * H_);

  for (int t = 0; t < T_; ++t) {
    const bf16_t* tok_t = toks_bf + (size_t)t * B_ * H_;

    // 1) sel + brl (shared A fragments): [tok|prev] @ {Wsel,Wbr}^T
    lin16_wmma<true><<<8, 128, 0, stream>>>(tok_t, prev_bf, nullptr, nullptr,
        Wsel_bf, bsel, sel_f, nullptr, 2 * H_, P_, ACT_NONE, (long)P_,
        Wbr_bf, bbr, brl_f);
    // 2) mixer (softmax / anchor / topk / normalize / topo)
    mixer_kernel<<<B_, 512, 0, stream>>>(sel_f, brl_f, positions, Wtp, btp,
        out_w, out_lm, out_bm, out_md, w_bf, topo_bf, t);
    // 3) patch = w @ patch_vals
    lin16_wmma<false><<<8, 128, 0, stream>>>(w_bf, nullptr, nullptr, nullptr,
        pv_bf, nullptr, nullptr, patch_bf, P_, H_, ACT_NONE, (long)H_,
        nullptr, nullptr, nullptr);
    // 4) t1 = gelu([tok|patch|prev] @ Wsb1^T + b)
    lin16_wmma<false><<<8, 128, 0, stream>>>(tok_t, patch_bf, prev_bf, nullptr,
        Wsb1_bf, bsb1, nullptr, t1_bf, 3 * H_, H_, ACT_GELU, (long)H_,
        nullptr, nullptr, nullptr);
    // 5) sec = tanh(t1 @ Wsb2^T + b)
    lin16_wmma<false><<<8, 128, 0, stream>>>(t1_bf, nullptr, nullptr, nullptr,
        Wsb2_bf, bsb2, sec_f, sec_bf, H_, H_, ACT_TANH, (long)H_,
        nullptr, nullptr, nullptr);
    // 6) phase routing (tiny)
    phase_kernel<<<1, 256, 0, stream>>>(sec_f, prev_f, Wpr, bpr, phase_embed,
        out_pw, ps_bf, t);
    // 7) t2 = gelu([sec|patch|ps|topo] @ Wsu1^T + b)
    lin16_wmma<false><<<8, 128, 0, stream>>>(sec_bf, patch_bf, ps_bf, topo_bf,
        Wsu1_bf, bsu1, nullptr, t2_bf, 4 * H_, H_, ACT_GELU, (long)H_,
        nullptr, nullptr, nullptr);
    // 8) su = tanh(t2 @ Wsu2^T + b)
    lin16_wmma<false><<<8, 128, 0, stream>>>(t2_bf, nullptr, nullptr, nullptr,
        Wsu2_bf, bsu2, su_f, su_bf, H_, H_, ACT_TANH, (long)H_,
        nullptr, nullptr, nullptr);
    // 9) g1 = gelu([su|prev] @ Wpg1^T + b)
    lin16_wmma<false><<<8, 128, 0, stream>>>(su_bf, prev_bf, nullptr, nullptr,
        Wpg1_bf, bpg1, g1_f, nullptr, 2 * H_, H_, ACT_GELU, (long)H_,
        nullptr, nullptr, nullptr);
    // 10) gate + hid + LayerNorm + clip, update prev
    gate_hid_kernel<<<B_, 512, 0, stream>>>(g1_f, su_f, Wpg2, bpg2, ln_g, ln_b,
        prev_f, prev_bf, hid_bf, out_hid, out_gate, t);
    // 11) logits = hid @ Wout^T, strided straight into d_out[b][t][v]
    lin16_wmma<false><<<128, 128, 0, stream>>>(hid_bf, nullptr, nullptr, nullptr,
        Wout_bf, nullptr, out_logits + (size_t)t * V_, nullptr,
        H_, V_, ACT_NONE, (long)T_ * V_, nullptr, nullptr, nullptr);
  }
}
